// BinHD_7430293422706
// MI455X (gfx1250) — 8-bit matrix operand order repaired and checked by reading, compiled, not run on hardware
//
#include <hip/hip_runtime.h>
#include <hip/hip_bf16.h>

#define BB 8192
#define FF 1024
#define DD 8192
#define CC 512

typedef __attribute__((ext_vector_type(2))) float v2f;
typedef __attribute__((ext_vector_type(8))) float v8f;
typedef __attribute__((ext_vector_type(8))) int   v8i;

__global__ void binhd_prep_hv(const float* __restrict__ hv,
                              signed char* __restrict__ hvS,
                              int* __restrict__ hvsum) {
    const int c = blockIdx.x;
    const int t = threadIdx.x;
    const float* row = hv + (size_t)c * DD;
    signed char* orow = hvS + (size_t)c * DD;
    int s = 0;
    for (int d = t; d < DD; d += 256) {
        int v = (int)row[d];
        orow[d] = (signed char)(1 - 2 * v);
        s += v;
    }
    __shared__ int red[256];
    red[t] = s;
    __syncthreads();
    for (int off = 128; off > 0; off >>= 1) {
        if (t < off) red[t] += red[t + off];
        __syncthreads();
    }
    if (t == 0) hvsum[c] = red[0];
}

__global__ void binhd_encode(const float* __restrict__ x,
                             const float* __restrict__ W,
                             unsigned char* __restrict__ enc8) {
    const int lane = threadIdx.x & 31;
    const int wave = threadIdx.x >> 5;
    const int m    = lane & 15;
    const int half = lane >> 4;

    const int rowBase = (blockIdx.y * 8 + wave) * 16;
    const int colBase = blockIdx.x * 64;

    v8f acc[4] = {};

    const float* __restrict__ xrow = x + (size_t)(rowBase + m) * FF;

    for (int k0 = 0; k0 < FF; k0 += 4) {
        const int ka = k0 + 2 * half;
        v2f a;
        a.x = xrow[ka];
        a.y = xrow[ka + 1];
        const float* __restrict__ w0 = W + (size_t)ka * DD + colBase + m;
        const float* __restrict__ w1 = w0 + DD;
#pragma unroll
        for (int t = 0; t < 4; ++t) {
            v2f b;
            b.x = w0[16 * t];
            b.y = w1[16 * t];
            acc[t] = __builtin_amdgcn_wmma_f32_16x16x4_f32(
                false, a, false, b, (short)0, acc[t], false, false);
        }
    }

#pragma unroll
    for (int t = 0; t < 4; ++t) {
#pragma unroll
        for (int e = 0; e < 8; ++e) {
            const int row = rowBase + 8 * half + e;
            const int col = colBase + 16 * t + m;
            enc8[(size_t)row * DD + col] = (acc[t][e] >= 0.0f) ? 1 : 0;
        }
    }
}

__global__ void binhd_response(const unsigned char* __restrict__ enc8,
                               const signed char* __restrict__ hvS,
                               const int* __restrict__ hvsum,
                               int* __restrict__ out) {
    const int lane = threadIdx.x & 31;
    const int wave = threadIdx.x >> 5;
    const int m    = lane & 15;
    const int half = lane >> 4;

    const int rowBase = blockIdx.y * 64;
    const int colW    = blockIdx.x * 128 + wave * 16;

    v8i acc[4] = {};

    const signed char* __restrict__ brow = hvS + (size_t)(colW + m) * DD;

    for (int k0 = 0; k0 < DD; k0 += 64) {
        const uint4 b0 = *(const uint4*)(brow + k0 + 16 * half);
        const uint4 b1 = *(const uint4*)(brow + k0 + 32 + 16 * half);
        const v8i bm = {(int)b0.x, (int)b0.y, (int)b0.z, (int)b0.w,
                        (int)b1.x, (int)b1.y, (int)b1.z, (int)b1.w};
#pragma unroll
        for (int r = 0; r < 4; ++r) {
            const unsigned char* __restrict__ abase =
                enc8 + (size_t)(rowBase + r * 16 + m) * DD + k0 + 16 * half;
            const uint2 a01 = *(const uint2*)(abase);
            const uint2 a23 = *(const uint2*)(abase + 8);
            const uint2 a45 = *(const uint2*)(abase + 32);
            const uint2 a67 = *(const uint2*)(abase + 40);
            const v8i am = {(int)a01.x, (int)a01.y, (int)a23.x, (int)a23.y,
                            (int)a45.x, (int)a45.y, (int)a67.x, (int)a67.y};
            acc[r] = __builtin_amdgcn_wmma_i32_16x16x64_iu8(
                false, am, true, bm, acc[r], false, false);
        }
    }

    const int bias = hvsum[colW + m];
#pragma unroll
    for (int r = 0; r < 4; ++r) {
#pragma unroll
        for (int e = 0; e < 8; ++e) {
            const int row = rowBase + r * 16 + 8 * half + e;
            out[(size_t)row * CC + colW + m] = acc[r][e] + bias;
        }
    }
}

extern "C" void kernel_launch(void* const* d_in, const int* in_sizes, int n_in,
                              void* d_out, int out_size, void* d_ws, size_t ws_size,
                              hipStream_t stream) {
    const float* x  = (const float*)d_in[0];
    const float* W  = (const float*)d_in[1];
    const float* hv = (const float*)d_in[2];
    int* out = (int*)d_out;

    unsigned char* enc8 = (unsigned char*)d_ws;
    signed char*   hvS  = (signed char*)((char*)d_ws + (size_t)BB * DD);
    int*           hvsm = (int*)((char*)d_ws + (size_t)BB * DD + (size_t)CC * DD);

    binhd_prep_hv<<<dim3(CC), dim3(256), 0, stream>>>(hv, hvS, hvsm);

    binhd_encode<<<dim3(DD / 64, BB / 128), dim3(256), 0, stream>>>(x, W, enc8);

    binhd_response<<<dim3(CC / 128, BB / 64), dim3(256), 0, stream>>>(
        enc8, hvS, hvsm, out);
}
